// Attention_1726576856724
// MI455X (gfx1250) — matrix operand order repaired and checked by reading, not compiled, not run on hardware
//
#include <hip/hip_runtime.h>

// ---------------- problem constants ----------------
constexpr int B_ = 4;
constexpr int N_ = 2048;
constexpr int C_ = 1024;
constexpr int H_ = 16;
constexpr int D_ = 64;   // head dim

typedef _Float16 f16x4 __attribute__((ext_vector_type(4)));
typedef _Float16 f16x8 __attribute__((ext_vector_type(8)));
typedef _Float16 v16h  __attribute__((ext_vector_type(16)));
typedef float    v8f   __attribute__((ext_vector_type(8)));

// ---------------- WMMA helper ----------------
__device__ __forceinline__ v8f wmma_f16(v16h a, v16h b, v8f c) {
  // D = A(16x32 f16) * B(32x16 f16) + C(16x16 f32)
  return __builtin_amdgcn_wmma_f32_16x16x32_f16(
      /*neg_a=*/false, a, /*neg_b=*/false, b,
      /*c_mod=*/(short)0, c, /*reuse_a=*/false, /*reuse_b=*/false);
}

// ---------------- CDNA5 async global->LDS copy (ASYNCcnt path) -----------
__device__ __forceinline__ void async_copy_b128(const _Float16* gsrc,
                                                _Float16* lds_dst) {
  const unsigned int laddr = (unsigned int)(uintptr_t)lds_dst; // LDS offset
  asm volatile("global_load_async_to_lds_b128 %0, %1, off"
               :: "v"(laddr), "v"(gsrc)
               : "memory");
}
__device__ __forceinline__ void wait_async0() {
  asm volatile("s_wait_asynccnt 0" ::: "memory");
}

// A-matrix fragment (16 rows x 32 K), row-major source with leading dim ld.
template <typename PT>
__device__ __forceinline__ v16h load_fragA(PT base, int ld) {
  const int lane = threadIdx.x & 31;
  const int r    = lane & 15;
  const int kb   = (lane >> 4) << 3;             // 0 or 8
  PT p = base + r * ld + kb;
  f16x8 lo = *(const f16x8*)p;                   // K = kb .. kb+7
  f16x8 hi = *(const f16x8*)(p + 16);            // K = kb+16 .. kb+23
  v16h f;
#pragma unroll
  for (int i = 0; i < 8; ++i) { f[i] = lo[i]; f[i + 8] = hi[i]; }
  return f;
}

// B-matrix fragment (32 K x 16 cols); element (k,n) = base[n*ld + k].
template <typename PT>
__device__ __forceinline__ v16h load_fragB(PT base, int ld) {
  const int lane = threadIdx.x & 31;
  const int n    = lane & 15;
  const int kb   = (lane >> 4) << 3;             // 0 or 8
  PT p = base + n * ld + kb;
  f16x8 lo = *(const f16x8*)p;                   // K = kb .. kb+7
  f16x8 hi = *(const f16x8*)(p + 16);            // K = kb+16 .. kb+23
  v16h f;
#pragma unroll
  for (int i = 0; i < 8; ++i) { f[i] = lo[i]; f[i + 8] = hi[i]; }
  return f;
}

// ---------------- f32 -> f16 conversion ----------------
__global__ void cvt_f32_f16(const float* __restrict__ in,
                            _Float16* __restrict__ out, int n) {
  int i = (blockIdx.x * blockDim.x + threadIdx.x) * 4;
  if (i + 3 < n) {
    float4 v = *(const float4*)(in + i);
    f16x4 o = { (_Float16)v.x, (_Float16)v.y, (_Float16)v.z, (_Float16)v.w };
    *(f16x4*)(out + i) = o;
  }
}

// ---------------- QKV projection -----------------------------------------
// Block = 256 threads (8 waves) computing a 128(M) x 64(N) tile.
// Shared W K-slice (64 cols x 32 K = 4KB) double-buffered in LDS via async
// DMA, software-pipelined with fixed buffer roles (no cur-phi).
__global__ void gemm_qkv(const _Float16* __restrict__ X,
                         const _Float16* __restrict__ W,
                         _Float16* __restrict__ Q,
                         _Float16* __restrict__ K,
                         _Float16* __restrict__ VT) {
  __shared__ __align__(16) _Float16 wtile[2][64 * 32];

  constexpr int NB = (3 * C_) / 64;              // 48 column blocks
  const int mb = blockIdx.x / NB;
  const int nb = blockIdx.x % NB;
  const int wave = threadIdx.x >> 5;
  const int m0 = mb * 128 + wave * 16;
  const int c0 = nb * 64;
  const int tid = threadIdx.x;
  const _Float16* Xp = X + m0 * C_;

  auto stage_w = [&](int buf, int k0) {
    const int row = tid >> 2;                    // 0..63 (output col)
    const int col = (tid & 3) << 3;              // 0,8,16,24 (K offset)
    async_copy_b128(W + (c0 + row) * C_ + k0 + col, &wtile[buf][row * 32 + col]);
  };

  v8f acc[4] = {v8f{}, v8f{}, v8f{}, v8f{}};

  auto compute = [&](const _Float16* kt, int k0) {
    __builtin_prefetch(Xp + k0 + 128, 0, 3);
    v16h a = load_fragA(Xp + k0, C_);
#pragma unroll
    for (int t = 0; t < 4; ++t)
      acc[t] = wmma_f16(a, load_fragB(kt + (t * 16) * 32, 32), acc[t]);
  };

  stage_w(0, 0);
  wait_async0();
  __syncthreads();

  for (int k0 = 0; k0 < C_; k0 += 64) {          // two 32-K halves per iter
    stage_w(1, k0 + 32);
    compute(&wtile[0][0], k0);
    wait_async0();
    __syncthreads();
    if (k0 + 64 < C_) stage_w(0, k0 + 64);
    compute(&wtile[1][0], k0 + 32);
    wait_async0();
    __syncthreads();
  }

  const int lane = threadIdx.x & 31;
  const int nloc = lane & 15;
  const int mhi  = (lane >> 4) << 3;
#pragma unroll
  for (int t = 0; t < 4; ++t) {
    const int c     = c0 + t * 16 + nloc;        // qkv channel in [0,3C)
    const int three = c >> 10;                   // 0=Q 1=K 2=V
    const int hd    = c & (C_ - 1);
    const int h     = hd >> 6;
    const int d     = hd & (D_ - 1);
#pragma unroll
    for (int v = 0; v < 8; ++v) {
      const int m    = m0 + mhi + v;             // global token row
      const int bb   = m >> 11;                  // / N_
      const int ntok = m & (N_ - 1);
      const int head = bb * H_ + h;
      const _Float16 val = (_Float16)acc[t][v];
      if (three == 0)      Q[(head * N_ + ntok) * D_ + d]  = val;
      else if (three == 1) K[(head * N_ + ntok) * D_ + d]  = val;
      else                 VT[(head * D_ + d) * N_ + ntok] = val;
    }
  }
}

// ---------------- flash attention ----------------
// 128 threads = 4 waves; each wave owns 16 query rows; block covers 64 rows.
// Shared K (32x64) + V^T (64x32) tile (8KB) double-buffered via async DMA,
// software-pipelined with fixed buffer roles.
__global__ void attn_kernel(const _Float16* __restrict__ Q,
                            const _Float16* __restrict__ K,
                            const _Float16* __restrict__ VT,
                            _Float16* __restrict__ O) {
  __shared__ __align__(16) _Float16 kv[2][4096];     // [K 32x64 | V 64x32]
  __shared__ __align__(16) _Float16 pbuf[4][16 * 32];

  const int wave = threadIdx.x >> 5;
  const int lane = threadIdx.x & 31;
  const int nloc = lane & 15;
  const int mhi  = (lane >> 4) << 3;
  const int tid  = threadIdx.x;

  const int qblk = blockIdx.x & 31;              // N/64 = 32 query blocks
  const int bh   = blockIdx.x >> 5;
  const int bb   = bh >> 4;
  const int h    = bh & (H_ - 1);
  const int head = bb * H_ + h;

  const _Float16* Qh = Q  + (size_t)head * N_ * D_;
  const _Float16* Kh = K  + (size_t)head * N_ * D_;
  const _Float16* Vh = VT + (size_t)head * D_ * N_;

  auto stage_kv = [&](int buf, int j0) {
#pragma unroll
    for (int i = 0; i < 4; ++i) {
      const int idx = (tid + i * 128) * 8;       // halves within 8KB tile
      if (idx < 2048) {                          // K part: [key][chan 64]
        const int row = idx >> 6, col = idx & 63;
        async_copy_b128(Kh + (j0 + row) * D_ + col, &kv[buf][idx]);
      } else {                                   // V part: [d][key 32]
        const int vi = idx - 2048;
        const int row = vi >> 5, col = vi & 31;
        async_copy_b128(Vh + row * N_ + j0 + col, &kv[buf][idx]);
      }
    }
  };

  const int q0 = qblk * 64 + wave * 16;
  const v16h qa0 = load_fragA(Qh + q0 * D_ + 0,  D_);
  const v16h qa1 = load_fragA(Qh + q0 * D_ + 32, D_);

  v8f o[4] = {v8f{}, v8f{}, v8f{}, v8f{}};
  float mrow[8], lrow[8];
#pragma unroll
  for (int v = 0; v < 8; ++v) { mrow[v] = -1e30f; lrow[v] = 0.0f; }

  const float scale = 0.125f;                    // 1/sqrt(64)
  _Float16* pb = &pbuf[wave][0];

  // one 32-key step: scores + online softmax + P@V, from a given LDS buffer
  auto process = [&](const _Float16* kt, const _Float16* vt) {
    v8f s0 = v8f{}, s1 = v8f{};
    s0 = wmma_f16(qa0, load_fragB(kt + 0 * 64 + 0,  64), s0);
    s0 = wmma_f16(qa1, load_fragB(kt + 0 * 64 + 32, 64), s0);
    s1 = wmma_f16(qa0, load_fragB(kt + 16 * 64 + 0,  64), s1);
    s1 = wmma_f16(qa1, load_fragB(kt + 16 * 64 + 32, 64), s1);

#pragma unroll
    for (int v = 0; v < 8; ++v) {
      float a = s0[v] * scale;
      float c = s1[v] * scale;
      float t = fmaxf(a, c);
      for (int off = 8; off; off >>= 1) t = fmaxf(t, __shfl_xor(t, off, 16));
      const float mn    = fmaxf(mrow[v], t);
      const float alpha = __expf(mrow[v] - mn);
      const float p0 = __expf(a - mn);
      const float p1 = __expf(c - mn);
      float rs = p0 + p1;
      for (int off = 8; off; off >>= 1) rs += __shfl_xor(rs, off, 16);
      lrow[v] = lrow[v] * alpha + rs;
      mrow[v] = mn;
      o[0][v] *= alpha; o[1][v] *= alpha; o[2][v] *= alpha; o[3][v] *= alpha;
      pb[(mhi + v) * 32 + nloc]      = (_Float16)p0;   // P transpose staging
      pb[(mhi + v) * 32 + nloc + 16] = (_Float16)p1;
    }

    const v16h pa = load_fragA((const _Float16*)pb, 32);
#pragma unroll
    for (int t = 0; t < 4; ++t)
      o[t] = wmma_f16(pa, load_fragB(vt + (t * 16) * 32, 32), o[t]);
  };

  stage_kv(0, 0);
  wait_async0();
  __syncthreads();

  for (int j0 = 0; j0 < N_; j0 += 64) {          // two 32-key halves per iter
    stage_kv(1, j0 + 32);
    process(&kv[0][0], &kv[0][2048]);
    wait_async0();
    __syncthreads();
    if (j0 + 64 < N_) stage_kv(0, j0 + 64);
    process(&kv[1][0], &kv[1][2048]);
    wait_async0();
    __syncthreads();
  }

  // normalize and store merged-head layout O[B*N, C] (f16)
#pragma unroll
  for (int t = 0; t < 4; ++t) {
#pragma unroll
    for (int v = 0; v < 8; ++v) {
      const int qtok = q0 + mhi + v;
      const int d    = t * 16 + nloc;
      O[(size_t)(bb * N_ + qtok) * C_ + h * D_ + d] =
          (_Float16)(o[t][v] / lrow[v]);
    }
  }
}

// ---------------- output projection: O @ Wproj^T + b ----------------
__global__ void gemm_proj(const _Float16* __restrict__ A,
                          const _Float16* __restrict__ W,
                          const float* __restrict__ bias,
                          float* __restrict__ out) {
  __shared__ __align__(16) _Float16 wtile[2][64 * 32];

  constexpr int NB = C_ / 64;                    // 16 column blocks
  const int mb = blockIdx.x / NB;
  const int nb = blockIdx.x % NB;
  const int wave = threadIdx.x >> 5;
  const int m0 = mb * 128 + wave * 16;
  const int c0 = nb * 64;
  const int tid = threadIdx.x;
  const _Float16* Ap = A + m0 * C_;

  auto stage_w = [&](int buf, int k0) {
    const int row = tid >> 2;
    const int col = (tid & 3) << 3;
    async_copy_b128(W + (c0 + row) * C_ + k0 + col, &wtile[buf][row * 32 + col]);
  };

  v8f acc[4] = {v8f{}, v8f{}, v8f{}, v8f{}};

  auto compute = [&](const _Float16* kt, int k0) {
    __builtin_prefetch(Ap + k0 + 128, 0, 3);
    v16h a = load_fragA(Ap + k0, C_);
#pragma unroll
    for (int t = 0; t < 4; ++t)
      acc[t] = wmma_f16(a, load_fragB(kt + (t * 16) * 32, 32), acc[t]);
  };

  stage_w(0, 0);
  wait_async0();
  __syncthreads();

  for (int k0 = 0; k0 < C_; k0 += 64) {
    stage_w(1, k0 + 32);
    compute(&wtile[0][0], k0);
    wait_async0();
    __syncthreads();
    if (k0 + 64 < C_) stage_w(0, k0 + 64);
    compute(&wtile[1][0], k0 + 32);
    wait_async0();
    __syncthreads();
  }

  const int lane = threadIdx.x & 31;
  const int nloc = lane & 15;
  const int mhi  = (lane >> 4) << 3;
#pragma unroll
  for (int t = 0; t < 4; ++t) {
    const int c = c0 + t * 16 + nloc;
    const float bz = bias[c];
#pragma unroll
    for (int v = 0; v < 8; ++v) {
      out[(size_t)(m0 + mhi + v) * C_ + c] = acc[t][v] + bz;
    }
  }
}

// ---------------- host-side launch ----------------
extern "C" void kernel_launch(void* const* d_in, const int* in_sizes, int n_in,
                              void* d_out, int out_size, void* d_ws, size_t ws_size,
                              hipStream_t stream) {
  const float* x     = (const float*)d_in[0];   // [B,N,C]
  const float* Wqkv  = (const float*)d_in[1];   // [3C,C]
  const float* Wproj = (const float*)d_in[2];   // [C,C]
  const float* bproj = (const float*)d_in[3];   // [C]
  float* out = (float*)d_out;

  char* ws = (char*)d_ws;
  size_t off = 0;
  auto carve = [&](size_t elems) {
    _Float16* p = (_Float16*)(ws + off);
    off += elems * sizeof(_Float16);
    return p;
  };
  const size_t tok = (size_t)B_ * N_ * C_;       // 8,388,608
  _Float16* x16  = carve(tok);
  _Float16* Wq16 = carve((size_t)3 * C_ * C_);
  _Float16* Wp16 = carve((size_t)C_ * C_);
  _Float16* q16  = carve(tok);                   // [B,H,N,D]
  _Float16* k16  = carve(tok);                   // [B,H,N,D]
  _Float16* vT16 = carve(tok);                   // [B,H,D,N]
  _Float16* o16  = carve(tok);                   // [B*N, C]

  // 1) precision conversion
  {
    int n = (int)tok;
    cvt_f32_f16<<<(n / 4 + 255) / 256, 256, 0, stream>>>(x, x16, n);
    n = 3 * C_ * C_;
    cvt_f32_f16<<<(n / 4 + 255) / 256, 256, 0, stream>>>(Wqkv, Wq16, n);
    n = C_ * C_;
    cvt_f32_f16<<<(n / 4 + 255) / 256, 256, 0, stream>>>(Wproj, Wp16, n);
  }

  // 2) QKV projection: (8192/128) * (3072/64) = 64*48 = 3072 blocks
  gemm_qkv<<<3072, 256, 0, stream>>>(x16, Wq16, q16, k16, vT16);

  // 3) flash attention: B*H*(N/64) = 2048 blocks of 4 waves
  attn_kernel<<<B_ * H_ * (N_ / 64), 128, 0, stream>>>(q16, k16, vT16, o16);

  // 4) output projection: (8192/128) * (1024/64) = 64*16 = 1024 blocks
  gemm_proj<<<1024, 256, 0, stream>>>(o16, Wp16, bproj, out);

  (void)in_sizes; (void)n_in; (void)out_size; (void)ws_size;
}
